// S4ConditionalModel_19301583028790
// MI455X (gfx1250) — matrix operand order repaired and checked by reading, not compiled, not run on hardware
//
#include <hip/hip_runtime.h>
#include <math.h>
#include <stdint.h>

// Problem constants (from reference)
#define BB 8
#define LL 65536
#define HH 32
#define NN 8
#define DEPTH 4
#define CLEN 256                 // l-chunk per workgroup
#define NCHUNK (LL / CLEN)       // 256 chunks per batch row
#define LPAD (CLEN + 1)          // padded LDS row stride (257 dwords -> conflict-free)

typedef __attribute__((ext_vector_type(16))) _Float16 v16h;
typedef __attribute__((ext_vector_type(8)))  float    v8f;
typedef unsigned int u32x4 __attribute__((ext_vector_type(4)));
typedef int          i32x8 __attribute__((ext_vector_type(8)));
typedef int          i32x4 __attribute__((ext_vector_type(4)));

// ---------------------------------------------------------------------------
// TDM: issue a 2D tile load (tile0 x tile1 elements of 4B) global->LDS with
// 1-dword padding after every 256 dwords (LDS row stride becomes 257 dwords).
// D# layout per CDNA5 ISA ch.8 (group0 128b, group1 256b; groups 2/3 zero).
// ---------------------------------------------------------------------------
__device__ __forceinline__ void tdm_load_tile_2d_padded(
    unsigned lds_addr, const void* gptr,
    unsigned tdim0, unsigned tdim1, unsigned tile0, unsigned tile1,
    unsigned stride0)
{
    unsigned long long ga = (unsigned long long)(uintptr_t)gptr;
    u32x4 g0;
    g0[0] = 1u;                                        // count=1, user descriptor
    g0[1] = lds_addr;                                  // LDS byte address
    g0[2] = (unsigned)(ga & 0xffffffffull);            // global addr [31:0]
    g0[3] = (unsigned)((ga >> 32) & 0x1ffffffull)      // global addr [56:32]
          | (2u << 30);                                // type = 2 ("image")
    i32x8 g1;
    g1[0] = (int)((2u << 16)                           // data_size = 4 bytes
                | (1u << 20)                           // pad_enable
                | (7u << 22)                           // pad_interval: 256 dwords
                | (0u << 25));                         // pad_amount: 1 dword
    g1[1] = (int)((tdim0 & 0xffffu) << 16);            // tensor_dim0[15:0]
    g1[2] = (int)((tdim0 >> 16) | ((tdim1 & 0xffffu) << 16));
    g1[3] = (int)((tdim1 >> 16) | (tile0 << 16));      // tile_dim0
    g1[4] = (int)(tile1 & 0xffffu);                    // tile_dim1 (tile_dim2 = 0)
    g1[5] = (int)stride0;                              // tensor_dim0_stride[31:0]
    g1[6] = 0;
    g1[7] = 0;
    i32x4 z4 = {0, 0, 0, 0};
#if defined(__clang_major__) && (__clang_major__ >= 23)
    i32x8 z8 = {0, 0, 0, 0, 0, 0, 0, 0};
    __builtin_amdgcn_tensor_load_to_lds(g0, g1, z4, z4, z8, 0);
#else
    __builtin_amdgcn_tensor_load_to_lds(g0, g1, z4, z4, 0);
#endif
}

// ---------------------------------------------------------------------------
// Kernel 0: conditioning MLP (2->16->32->32, ReLU) + FiLM projection per depth
// ---------------------------------------------------------------------------
__global__ void cond_film_kernel(const float* __restrict__ params,
                                 const float* __restrict__ W1, const float* __restrict__ b1,
                                 const float* __restrict__ W2, const float* __restrict__ b2,
                                 const float* __restrict__ W3, const float* __restrict__ b3,
                                 const float* __restrict__ filmW, const float* __restrict__ filmB,
                                 float* __restrict__ filmws)
{
    int t = threadIdx.x;
    if (t >= BB) return;
    float p0 = params[t * 2 + 0], p1 = params[t * 2 + 1];
    float c1[16];
#pragma unroll
    for (int i = 0; i < 16; ++i)
        c1[i] = fmaxf(W1[i * 2] * p0 + W1[i * 2 + 1] * p1 + b1[i], 0.f);
    float c2[32];
    for (int i = 0; i < 32; ++i) {
        float s = b2[i];
#pragma unroll
        for (int j = 0; j < 16; ++j) s += W2[i * 16 + j] * c1[j];
        c2[i] = fmaxf(s, 0.f);
    }
    float c3[32];
    for (int i = 0; i < 32; ++i) {
        float s = b3[i];
#pragma unroll
        for (int j = 0; j < 32; ++j) s += W3[i * 32 + j] * c2[j];
        c3[i] = fmaxf(s, 0.f);
    }
    for (int d = 0; d < DEPTH; ++d)
        for (int o = 0; o < 2 * HH; ++o) {
            float s = filmB[d * 2 * HH + o];
#pragma unroll
            for (int j = 0; j < 32; ++j) s += filmW[(d * 2 * HH + o) * 32 + j] * c3[j];
            filmws[(d * BB + t) * 2 * HH + o] = s;
        }
}

// ---------------------------------------------------------------------------
// Kernel 1: expand  h[b,h,l] = x[b,l]*W[h] + b[h]
// ---------------------------------------------------------------------------
__global__ __launch_bounds__(256) void expand_kernel(const float* __restrict__ x,
                                                     const float* __restrict__ eW,
                                                     const float* __restrict__ eb,
                                                     float* __restrict__ hws)
{
    size_t idx = (size_t)blockIdx.x * blockDim.x + threadIdx.x;
    int l  = (int)(idx % LL);
    int bh = (int)(idx / LL);
    int h  = bh % HH;
    int b  = bh / HH;
    hws[idx] = x[(size_t)b * LL + l] * eW[h] + eb[h];
}

// ---------------------------------------------------------------------------
// Kernel 2 (per depth): TDM chunk stage -> WMMA channel mix + ReLU -> local
// S4D scan over the chunk (zero initial state); chunk-final states -> Sloc.
// ---------------------------------------------------------------------------
__global__ __launch_bounds__(256) void mix_scan_kernel(
    const float* __restrict__ hws, float* __restrict__ uws, float2* __restrict__ Sloc,
    const float* __restrict__ linW, const float* __restrict__ linb,
    const float* __restrict__ log_dt, const float* __restrict__ log_A_real,
    const float* __restrict__ A_imag, const float* __restrict__ C_re,
    const float* __restrict__ C_im, const float* __restrict__ D_skip, int d)
{
    __shared__ float su[HH * LPAD];                 // 33 KB padded: h-chunk, then u, then y
    const int ck = blockIdx.x, b = blockIdx.y;
    const int t = threadIdx.x;
    const int wave = t >> 5, lane = t & 31;
    const int m15 = lane & 15, hi = lane >> 4;

    // ---- Stage A0: async TDM load of the 32x256 f32 tile into padded LDS
    if (t < 32) {
        tdm_load_tile_2d_padded((unsigned)(uintptr_t)su,
                                hws + ((size_t)b * HH) * LL + (size_t)ck * CLEN,
                                /*tensor_dim0*/ (unsigned)LL, /*tensor_dim1*/ HH,
                                /*tile0*/ CLEN, /*tile1*/ HH,
                                /*stride0*/ (unsigned)LL);
        __builtin_amdgcn_s_wait_tensorcnt(0);
    }
    __syncthreads();

    // ---- Stage A1: fragments.  A: W tiles from global (L2-hot).  B: from LDS,
// lane holds column l = lane&15, K in table order (8*hi + i, 16 + 8*hi + i-8).
    v16h afrag[2], bfrag[2];
    {
        __builtin_prefetch(linW + (size_t)d * HH * HH, 0, 0);   // global_prefetch_b8
#pragma unroll
        for (int gt = 0; gt < 2; ++gt) {
            const float* wrow = linW + ((size_t)d * HH + (gt * 16 + m15)) * HH;
#pragma unroll
            for (int r = 0; r < 8; ++r) {
                int k = ((r < 4) ? 0 : 16) + hi * 8 + 2 * (r & 3);
                afrag[gt][2 * r]     = (_Float16)wrow[k];
                afrag[gt][2 * r + 1] = (_Float16)wrow[k + 1];
            }
        }
#pragma unroll
        for (int q = 0; q < 2; ++q) {
            int lt = wave * 2 + q;
            const float* hb = &su[lt * 16 + m15];
#pragma unroll
            for (int i = 0; i < 16; ++i) bfrag[q][i] = (_Float16)hb[(((i < 8) ? 0 : 16) + hi * 8 + (i & 7)) * LPAD];
        }
    }
    __syncthreads();                                // everyone done reading h-chunk

    // ---- Stage A2: out[g,l] = relu(W[g,:]*h[:,l] + b[g]) via v_wmma, into su
    {
#pragma unroll
        for (int q = 0; q < 2; ++q) {
            int lt = wave * 2 + q;
#pragma unroll
            for (int gt = 0; gt < 2; ++gt) {
                v8f acc = {};
                acc = __builtin_amdgcn_wmma_f32_16x16x32_f16(
                          false, afrag[gt], false, bfrag[q], (short)0, acc, false, false);
#pragma unroll
                for (int r = 0; r < 8; ++r) {       // D: VGPR r -> M = r + 8*hi, N = lane&15
                    int g = gt * 16 + r + 8 * hi;
                    float v = acc[r] + linb[d * HH + g];
                    su[g * LPAD + lt * 16 + m15] = fmaxf(v, 0.f);
                }
            }
        }
    }
    __syncthreads();

    // ---- Stage B: per-(h,n) diagonal recurrence, shfl reduce over n (wave32)
    {
        const int h = t >> 3, n = t & 7;            // 8 lanes per h, same wave
        const int pi = (d * HH + h) * NN + n;
        float dtv = __expf(log_dt[d * HH + h]);
        float Are = -__expf(log_A_real[pi]);
        float Aim = A_imag[pi];
        float ar = dtv * Are, ai = dtv * Aim;
        float er = __expf(ar);
        float lr = er * __cosf(ai), li = er * __sinf(ai);      // lambda = exp(dt*A)
        // Cd = (Cre + i Cim) * (lambda - 1)/A
        float den = Are * Are + Aim * Aim;
        float xr = lr - 1.f, xi = li;
        float wr = (xr * Are + xi * Aim) / den, wi = (xi * Are - xr * Aim) / den;
        float Cre = C_re[pi], Cim = C_im[pi];
        float cdr = Cre * wr - Cim * wi, cdi = Cre * wi + Cim * wr;
        float dsk = D_skip[d * HH + h];
        float sr = 0.f, si = 0.f;
        for (int l = 0; l < CLEN; ++l) {
            float u = su[h * LPAD + l];             // LDS broadcast to the 8-lane group
            float nr = lr * sr - li * si + u;
            si = lr * si + li * sr;
            sr = nr;
            float contrib = 2.f * (cdr * sr - cdi * si);
            contrib += __shfl_xor(contrib, 1, 32);
            contrib += __shfl_xor(contrib, 2, 32);
            contrib += __shfl_xor(contrib, 4, 32);
            if (n == 0) su[h * LPAD + l] = contrib + dsk * u;   // in place: u consumed
        }
        Sloc[((size_t)(b * NCHUNK + ck) * HH + h) * NN + n] = make_float2(sr, si);
    }
    __syncthreads();

    // ---- Stage C: coalesced writeback of y_partial
    {
        float* ub = uws + ((size_t)b * HH) * LL + (size_t)ck * CLEN;
#pragma unroll
        for (int i = 0; i < HH; ++i)
            ub[(size_t)i * LL + t] = su[i * LPAD + t];
    }
}

// ---------------------------------------------------------------------------
// Kernel 3 (per depth): carry scan over chunks: Sin[c] = Sg[c-1],
//   Sg[c] = lambda^CLEN * Sg[c-1] + Sloc[c]   (lambda^CLEN closed-form)
// ---------------------------------------------------------------------------
__global__ __launch_bounds__(256) void state_scan_kernel(
    const float* __restrict__ log_dt, const float* __restrict__ log_A_real,
    const float* __restrict__ A_imag, const float2* __restrict__ Sloc,
    float2* __restrict__ Sin, int d)
{
    int b = blockIdx.x, t = threadIdx.x;
    int h = t >> 3, n = t & 7;
    int pi = (d * HH + h) * NN + n;
    float dtv = __expf(log_dt[d * HH + h]);
    float Are = -__expf(log_A_real[pi]);
    float Aim = A_imag[pi];
    float ar = dtv * Are * (float)CLEN, ai = dtv * Aim * (float)CLEN;
    float er = __expf(ar);
    float Lr = er * __cosf(ai), Li = er * __sinf(ai);
    float sr = 0.f, si = 0.f;
    for (int c = 0; c < NCHUNK; ++c) {
        size_t idx = ((size_t)(b * NCHUNK + c) * HH + h) * NN + n;
        Sin[idx] = make_float2(sr, si);
        float2 sl = Sloc[idx];
        float nr = Lr * sr - Li * si + sl.x;
        si = Lr * si + Li * sr + sl.y;
        sr = nr;
    }
}

// ---------------------------------------------------------------------------
// Kernel 4 (per depth): state correction + FiLM + ReLU + depthwise residual
//   y[l] = y_partial[l] + 2 Re(sum_n Cd * lambda^(l_local+1) * Sin)
//   h    = relu(y*g + beta) + res_w * h
// ---------------------------------------------------------------------------
__global__ __launch_bounds__(256) void film_res_kernel(
    const float* __restrict__ log_dt, const float* __restrict__ log_A_real,
    const float* __restrict__ A_imag, const float* __restrict__ C_re,
    const float* __restrict__ C_im, const float* __restrict__ res_w,
    const float* __restrict__ filmws, const float2* __restrict__ Sin,
    const float* __restrict__ uws, float* __restrict__ hws, int d)
{
    const int ck = blockIdx.x, b = blockIdx.y;
    const int t = threadIdx.x;
    const int h = t >> 3, sl = t & 7;
    const int SEG = CLEN / 8;                       // 32 l's per thread
    const int l0 = sl * SEG;

    float dtv = __expf(log_dt[d * HH + h]);
    float g   = filmws[(d * BB + b) * 2 * HH + h];
    float be  = filmws[(d * BB + b) * 2 * HH + HH + h];
    float rw  = res_w[d * HH + h];

    float lr8[NN], li8[NN], cdr[NN], cdi[NN], qr[NN], qi[NN];
#pragma unroll
    for (int n = 0; n < NN; ++n) {
        int pi = (d * HH + h) * NN + n;
        float Are = -__expf(log_A_real[pi]);
        float Aim = A_imag[pi];
        float ar = dtv * Are, ai = dtv * Aim;
        float er = __expf(ar);
        float lr = er * __cosf(ai), li = er * __sinf(ai);
        float den = Are * Are + Aim * Aim;
        float xr = lr - 1.f, xi = li;
        float wr = (xr * Are + xi * Aim) / den, wi = (xi * Are - xr * Aim) / den;
        float Cre = C_re[pi], Cim = C_im[pi];
        cdr[n] = Cre * wr - Cim * wi;
        cdi[n] = Cre * wi + Cim * wr;
        lr8[n] = lr; li8[n] = li;
        float2 sv = Sin[((size_t)(b * NCHUNK + ck) * HH + h) * NN + n];
        float k  = (float)(l0 + 1);
        float pw = __expf(ar * k);
        float pr = pw * __cosf(ai * k), pm = pw * __sinf(ai * k);   // lambda^(l0+1)
        qr[n] = pr * sv.x - pm * sv.y;
        qi[n] = pr * sv.y + pm * sv.x;
    }
    size_t base = ((size_t)(b * HH + h)) * LL + (size_t)ck * CLEN + l0;
    for (int j = 0; j < SEG; ++j) {
        float corr = 0.f;
#pragma unroll
        for (int n = 0; n < NN; ++n) corr += 2.f * (cdr[n] * qr[n] - cdi[n] * qi[n]);
        float y = uws[base + j] + corr;
        float z = fmaxf(y * g + be, 0.f);
        float ho = hws[base + j];
        hws[base + j] = z + rw * ho;
#pragma unroll
        for (int n = 0; n < NN; ++n) {
            float nr = lr8[n] * qr[n] - li8[n] * qi[n];
            qi[n] = lr8[n] * qi[n] + li8[n] * qr[n];
            qr[n] = nr;
        }
    }
}

// ---------------------------------------------------------------------------
// Kernel 5: contract + tanh + side chain
// ---------------------------------------------------------------------------
__global__ __launch_bounds__(256) void contract_kernel(const float* __restrict__ x,
                                                       const float* __restrict__ hws,
                                                       const float* __restrict__ cW,
                                                       const float* __restrict__ cb,
                                                       float* __restrict__ out)
{
    size_t idx = (size_t)blockIdx.x * blockDim.x + threadIdx.x;
    int l = (int)(idx % LL);
    int b = (int)(idx / LL);
    float acc = cb[0];
#pragma unroll
    for (int h = 0; h < HH; ++h)
        acc += hws[((size_t)(b * HH + h)) * LL + l] * cW[h];
    out[idx] = x[idx] * tanhf(acc);
}

// ---------------------------------------------------------------------------
extern "C" void kernel_launch(void* const* d_in, const int* in_sizes, int n_in,
                              void* d_out, int out_size, void* d_ws, size_t ws_size,
                              hipStream_t stream)
{
    const float* x      = (const float*)d_in[0];
    const float* params = (const float*)d_in[1];
    const float* W1     = (const float*)d_in[2];
    const float* b1     = (const float*)d_in[3];
    const float* W2     = (const float*)d_in[4];
    const float* b2     = (const float*)d_in[5];
    const float* W3     = (const float*)d_in[6];
    const float* b3     = (const float*)d_in[7];
    const float* eW     = (const float*)d_in[8];
    const float* eb     = (const float*)d_in[9];
    const float* linW   = (const float*)d_in[10];
    const float* linb   = (const float*)d_in[11];
    const float* log_dt = (const float*)d_in[12];
    const float* logAr  = (const float*)d_in[13];
    const float* Aim    = (const float*)d_in[14];
    const float* Cre    = (const float*)d_in[15];
    const float* Cim    = (const float*)d_in[16];
    const float* Dsk    = (const float*)d_in[17];
    const float* filmW  = (const float*)d_in[18];
    const float* filmB  = (const float*)d_in[19];
    const float* resw   = (const float*)d_in[20];
    const float* cW     = (const float*)d_in[21];
    const float* cb     = (const float*)d_in[22];
    float* out = (float*)d_out;

    // Workspace layout (floats)
    const size_t HLB = (size_t)BB * HH * LL;                    // 16,777,216
    const size_t STF = (size_t)BB * NCHUNK * HH * NN * 2;       // 1,048,576
    float*  ws     = (float*)d_ws;
    float*  hws    = ws;                                        // h  [B,H,L]
    float*  uws    = ws + HLB;                                  // y_partial [B,H,L]
    float2* sloc   = (float2*)(ws + 2 * HLB);                   // chunk-final states
    float2* sinS   = (float2*)(ws + 2 * HLB + STF);             // incoming states
    float*  filmws = ws + 2 * HLB + 2 * STF;                    // FiLM g/beta [D,B,2H]

    cond_film_kernel<<<1, 64, 0, stream>>>(params, W1, b1, W2, b2, W3, b3,
                                           filmW, filmB, filmws);
    expand_kernel<<<(int)(HLB / 256), 256, 0, stream>>>(x, eW, eb, hws);

    for (int d = 0; d < DEPTH; ++d) {
        mix_scan_kernel<<<dim3(NCHUNK, BB), 256, 0, stream>>>(
            hws, uws, sloc, linW, linb, log_dt, logAr, Aim, Cre, Cim, Dsk, d);
        state_scan_kernel<<<BB, 256, 0, stream>>>(log_dt, logAr, Aim, sloc, sinS, d);
        film_res_kernel<<<dim3(NCHUNK, BB), 256, 0, stream>>>(
            log_dt, logAr, Aim, Cre, Cim, resw, filmws, sinS, uws, hws, d);
    }
    contract_kernel<<<(int)(((size_t)BB * LL) / 256), 256, 0, stream>>>(x, hws, cW, cb, out);
}
